// GAT_relaevance_53283364274283
// MI455X (gfx1250) — matrix operand order repaired and checked by reading, not compiled, not run on hardware
//
#include <hip/hip_runtime.h>
#include <hip/hip_bf16.h>

// ---------------------------------------------------------------------------
// GAT (2-layer, 3-head) + MLP head forward for MI455X (gfx1250).
// Dense projections: v_wmma_f32_16x16x32_f16, fully unrolled over K,
// B panel staged via GLOBAL_LOAD_ASYNC_TO_LDS (when available).
// Edge softmax / aggregation: float atomics + global_prefetch.
// ---------------------------------------------------------------------------

static constexpr int IN_F = 128;
static constexpr int HID  = 64;
static constexpr int NH   = 3;
static constexpr int F    = NH * HID; // 192
static constexpr int OUTF = 16;
static constexpr float NEG_SLOPE = 0.2f;

typedef __attribute__((ext_vector_type(16))) _Float16 v16h;
typedef __attribute__((ext_vector_type(8)))  float    v8f;

typedef __attribute__((address_space(1))) int* gint_p;   // global int*
typedef __attribute__((address_space(3))) int* lint_p;   // LDS int*

#if defined(__HIP_DEVICE_COMPILE__) && __has_builtin(__builtin_amdgcn_global_load_async_to_lds_b32)
#define HAVE_ASYNC_LDS 1
#else
#define HAVE_ASYNC_LDS 0
#endif

// ---------------------------------------------------------------------------
// WMMA GEMM: Y(N x NCOLS) = X(N x K) * W(K x NCOLS) [+ bias][relu]
// block = 160 threads = 5 waves; each wave -> one 16x16 output tile.
// Block tile: 80 rows x 16 cols. K % 32 == 0, NCOLS % 16 == 0 (compile-time).
// ---------------------------------------------------------------------------
template <int K, int NCOLS>
__global__ void __launch_bounds__(160)
gemm_wmma_f16(const float* __restrict__ X, const float* __restrict__ W,
              float* __restrict__ Y, int Mrows,
              const float* __restrict__ bias, int do_relu)
{
    constexpr int KSTEPS = K / 32;
    __shared__ __align__(16) float    ldsW[16 * K];          // [k][col] f32 staging
    __shared__ __align__(32) _Float16 ldsB[16 * K];          // [step][col][32] fragments

    const int lane = threadIdx.x & 31;
    const int wave = threadIdx.x >> 5;
    const int col0 = blockIdx.x * 16;
    int row0 = blockIdx.y * (16 * 5) + wave * 16;
    if (row0 + 16 > Mrows) row0 = Mrows - 16;   // wave-uniform clamp

    const int  r  = lane & 15;
    const bool hi = (lane & 16) != 0;

    // ---- stage B panel (K x 16 f32) into LDS --------------------------------
#if HAVE_ASYNC_LDS
    for (int i = threadIdx.x; i < 16 * K; i += 160) {
        int k = i >> 4, c = i & 15;
        float* gp = const_cast<float*>(W) + k * NCOLS + col0 + c;
        __builtin_amdgcn_global_load_async_to_lds_b32(
            (gint_p)gp, (lint_p)&ldsW[i], 0, 0);
    }
#if __has_builtin(__builtin_amdgcn_s_wait_asynccnt)
    __builtin_amdgcn_s_wait_asynccnt(0);
#else
    asm volatile("s_wait_asynccnt 0" ::: "memory");
#endif
#else
    for (int i = threadIdx.x; i < 16 * K; i += 160) {
        int k = i >> 4, c = i & 15;
        ldsW[i] = W[k * NCOLS + col0 + c];
    }
#endif
    __syncthreads();

    // ---- convert to f16 in WMMA B-fragment layout ---------------------------
    // ldsB[step*512 + c*32 + j] = (f16) W[k0=step*32+j][col0+c]
    for (int i = threadIdx.x; i < 512 * KSTEPS; i += 160) {
        int step = i >> 9, rem = i & 511;
        int c = rem >> 5, j = rem & 31;
        ldsB[i] = (_Float16)ldsW[(step * 32 + j) * 16 + c];
    }
    __syncthreads();

    // ---- fully unrolled WMMA chain ------------------------------------------
    const float*    xrow = X + (size_t)(row0 + r) * K + (hi ? 16 : 0);
    const _Float16* brow = &ldsB[r * 32 + (hi ? 16 : 0)];

    v8f acc = {0.f, 0.f, 0.f, 0.f, 0.f, 0.f, 0.f, 0.f};
#pragma unroll
    for (int step = 0; step < KSTEPS; ++step) {
        // A fragment (16x32 f16): lane<16 -> K {0..15}; lane>=16 -> {16..31}
        const float4* xp = reinterpret_cast<const float4*>(xrow + step * 32);
        float4 u0 = xp[0], u1 = xp[1], u2 = xp[2], u3 = xp[3];
        v16h a = { (_Float16)u0.x, (_Float16)u0.y, (_Float16)u0.z, (_Float16)u0.w,
                   (_Float16)u1.x, (_Float16)u1.y, (_Float16)u1.z, (_Float16)u1.w,
                   (_Float16)u2.x, (_Float16)u2.y, (_Float16)u2.z, (_Float16)u2.w,
                   (_Float16)u3.x, (_Float16)u3.y, (_Float16)u3.z, (_Float16)u3.w };
        v16h b = *reinterpret_cast<const v16h*>(brow + step * 512);
        acc = __builtin_amdgcn_wmma_f32_16x16x32_f16(
                  false, a, false, b, (short)0, acc, false, false);
    }

    // C/D layout: VGPR i -> row (row0 + i + (hi?8:0)), col = col0 + (lane&15)
    const int col = col0 + r;
    const float bv = bias ? bias[col] : 0.f;
    const int rbase = row0 + (hi ? 8 : 0);
#pragma unroll
    for (int i = 0; i < 8; ++i) {
        float v = acc[i] + bv;
        if (do_relu) v = v > 0.f ? v : 0.f;
        Y[(size_t)(rbase + i) * NCOLS + col] = v;
    }
}

// ---------------------------------------------------------------------------
// el[n,h] = <feat[n,h,:], al[h,:]> ; er likewise
// ---------------------------------------------------------------------------
__global__ void node_scores_k(const float* __restrict__ feat,
                              const float* __restrict__ al,
                              const float* __restrict__ ar,
                              float* __restrict__ el, float* __restrict__ er,
                              int total)
{
    int t = blockIdx.x * blockDim.x + threadIdx.x;
    if (t >= total) return;
    int node = t / NH, h = t - node * NH;
    const float* fp = feat + (size_t)node * F + h * HID;
    const float* ap = al + h * HID;
    const float* rp = ar + h * HID;
    float sl = 0.f, sr = 0.f;
    for (int d = 0; d < HID; d += 4) {
        float4 f = *(const float4*)(fp + d);
        float4 a = *(const float4*)(ap + d);
        float4 g = *(const float4*)(rp + d);
        sl += f.x * a.x + f.y * a.y + f.z * a.z + f.w * a.w;
        sr += f.x * g.x + f.y * g.y + f.z * g.z + f.w * g.w;
    }
    el[t] = sl; er[t] = sr;
}

// Order-preserving uint encoding of float (atomicMax over signed floats)
__device__ __forceinline__ unsigned encf(float f) {
    unsigned u = __float_as_uint(f);
    return (u & 0x80000000u) ? ~u : (u | 0x80000000u);
}
__device__ __forceinline__ float decf(unsigned u) {
    u = (u & 0x80000000u) ? (u & 0x7fffffffu) : ~u;
    return __uint_as_float(u);
}
static constexpr unsigned ENC_NEG_INF = 0x007FFFFFu; // encf(-inf)

__global__ void edge_max_k(const int* __restrict__ src, const int* __restrict__ dst,
                           const float* __restrict__ el, const float* __restrict__ er,
                           unsigned* __restrict__ mb, int Ne)
{
    int e = blockIdx.x * blockDim.x + threadIdx.x;
    if (e >= Ne) return;
    int s = src[e], d = dst[e];
#pragma unroll
    for (int h = 0; h < NH; ++h) {
        float sc = el[s * NH + h] + er[d * NH + h];
        sc = sc > 0.f ? sc : NEG_SLOPE * sc;
        atomicMax(&mb[d * NH + h], encf(sc));
    }
}

__global__ void edge_expsum_k(const int* __restrict__ src, const int* __restrict__ dst,
                              const float* __restrict__ el, const float* __restrict__ er,
                              const unsigned* __restrict__ mb,
                              float* __restrict__ exb, float* __restrict__ sb,
                              const float* __restrict__ feat, int Ne)
{
    int e = blockIdx.x * blockDim.x + threadIdx.x;
    if (e >= Ne) return;
    int s = src[e], d = dst[e];
    // warm L2 with the src feature row for the upcoming aggregation pass
    __builtin_prefetch(feat + (size_t)s * F, 0, 3);
#pragma unroll
    for (int h = 0; h < NH; ++h) {
        float sc = el[s * NH + h] + er[d * NH + h];
        sc = sc > 0.f ? sc : NEG_SLOPE * sc;
        float ex = __expf(sc - decf(mb[d * NH + h]));
        exb[e * NH + h] = ex;
        atomicAdd(&sb[d * NH + h], ex);
    }
}

// one thread per (edge, 4-float chunk of the 192-wide feature row)
__global__ void edge_agg_k(const int* __restrict__ src, const int* __restrict__ dst,
                           const float* __restrict__ feat,
                           const float* __restrict__ exb, const float* __restrict__ sb,
                           float* __restrict__ agg, int Ne)
{
    int t = blockIdx.x * blockDim.x + threadIdx.x;
    int total = Ne * (F / 4);
    if (t >= total) return;
    int e = t / (F / 4), q = t - e * (F / 4);
    int s = src[e], d = dst[e];
    int h = q >> 4;
    float alpha = exb[e * NH + h] / sb[d * NH + h];
    int idx = q * 4;
    float4 f = *(const float4*)(feat + (size_t)s * F + idx);
    float* op = agg + (size_t)d * F + idx;
    atomicAdd(op + 0, alpha * f.x);
    atomicAdd(op + 1, alpha * f.y);
    atomicAdd(op + 2, alpha * f.z);
    atomicAdd(op + 3, alpha * f.w);
}

__global__ void bias_relu_k(float* __restrict__ x, const float* __restrict__ b, int total)
{
    int t = blockIdx.x * blockDim.x + threadIdx.x;
    if (t >= total) return;
    int idx = t % F;
    float v = x[t] + b[idx];
    x[t] = v > 0.f ? v : 0.f;
}

__global__ void fill_f_k(float* __restrict__ p, float v, int n)
{
    int t = blockIdx.x * blockDim.x + threadIdx.x;
    if (t < n) p[t] = v;
}
__global__ void fill_u_k(unsigned* __restrict__ p, unsigned v, int n)
{
    int t = blockIdx.x * blockDim.x + threadIdx.x;
    if (t < n) p[t] = v;
}

// ---------------------------------------------------------------------------
extern "C" void kernel_launch(void* const* d_in, const int* in_sizes, int n_in,
                              void* d_out, int out_size, void* d_ws, size_t ws_size,
                              hipStream_t stream)
{
    const float* features = (const float*)d_in[0];
    const int*   src      = (const int*)  d_in[1];
    const int*   dst      = (const int*)  d_in[2];
    const float* W1  = (const float*)d_in[3];
    const float* al1 = (const float*)d_in[4];
    const float* ar1 = (const float*)d_in[5];
    const float* b1  = (const float*)d_in[6];
    const float* W2  = (const float*)d_in[7];
    const float* al2 = (const float*)d_in[8];
    const float* ar2 = (const float*)d_in[9];
    const float* b2  = (const float*)d_in[10];
    const float* w3  = (const float*)d_in[11];
    const float* b3  = (const float*)d_in[12];
    const float* w4  = (const float*)d_in[13];
    const float* b4  = (const float*)d_in[14];
    float* out = (float*)d_out;

    const int Nn = in_sizes[0] / IN_F;   // 100000 nodes
    const int Ne = in_sizes[1];          // 1600000 edges

    // workspace layout (floats)
    float*    feat = (float*)d_ws;                        // N*F
    float*    agg  = feat + (size_t)Nn * F;               // N*F
    float*    el   = agg  + (size_t)Nn * F;               // N*NH
    float*    er   = el   + (size_t)Nn * NH;              // N*NH
    unsigned* mb   = (unsigned*)(er + (size_t)Nn * NH);   // N*NH
    float*    sb   = (float*)(mb + (size_t)Nn * NH);      // N*NH
    float*    exb  = sb   + (size_t)Nn * NH;              // E*NH
    float*    hb   = exb  + (size_t)Ne * NH;              // N*HID

    const int TPB = 256;
    auto cdiv = [](long a, long b) { return (int)((a + b - 1) / b); };
    const int rowBlocks = cdiv(Nn, 80);

    auto run_edges = [&](const float* al_, const float* ar_, const float* bias) {
        node_scores_k<<<cdiv((long)Nn * NH, TPB), TPB, 0, stream>>>(
            feat, al_, ar_, el, er, Nn * NH);
        fill_u_k<<<cdiv((long)Nn * NH, TPB), TPB, 0, stream>>>(mb, ENC_NEG_INF, Nn * NH);
        fill_f_k<<<cdiv((long)Nn * NH, TPB), TPB, 0, stream>>>(sb, 0.f, Nn * NH);
        fill_f_k<<<cdiv((long)Nn * F, TPB), TPB, 0, stream>>>(agg, 0.f, Nn * F);
        edge_max_k<<<cdiv(Ne, TPB), TPB, 0, stream>>>(src, dst, el, er, mb, Ne);
        edge_expsum_k<<<cdiv(Ne, TPB), TPB, 0, stream>>>(src, dst, el, er, mb, exb, sb, feat, Ne);
        edge_agg_k<<<cdiv((long)Ne * (F / 4), TPB), TPB, 0, stream>>>(
            src, dst, feat, exb, sb, agg, Ne);
        bias_relu_k<<<cdiv((long)Nn * F, TPB), TPB, 0, stream>>>(agg, bias, Nn * F);
    };

    // layer 1: features (N x 128) -> agg = x1 (N x 192)
    gemm_wmma_f16<IN_F, F><<<dim3(F / 16, rowBlocks), 160, 0, stream>>>(
        features, W1, feat, Nn, nullptr, 0);
    run_edges(al1, ar1, b1);

    // layer 2: x1 (N x 192) -> agg = x2 (N x 192)
    gemm_wmma_f16<F, F><<<dim3(F / 16, rowBlocks), 160, 0, stream>>>(
        agg, W2, feat, Nn, nullptr, 0);
    run_edges(al2, ar2, b2);

    // MLP head: h = relu(x2 @ w3 + b3) ; out = h @ w4 + b4
    gemm_wmma_f16<F, HID><<<dim3(HID / 16, rowBlocks), 160, 0, stream>>>(
        agg, w3, hb, Nn, b3, 1);
    gemm_wmma_f16<HID, OUTF><<<dim3(OUTF / 16, rowBlocks), 160, 0, stream>>>(
        hb, w4, out, Nn, b4, 0);
}
